// EncoderLayer_85761906967053
// MI455X (gfx1250) — matrix operand order repaired and checked by reading, not compiled, not run on hardware
//
#include <hip/hip_runtime.h>

typedef unsigned int u32;
typedef unsigned short u16;
typedef unsigned long long u64;

typedef __attribute__((ext_vector_type(16))) __bf16 v16bf;
typedef __attribute__((ext_vector_type(8)))  float  v8f;
typedef __attribute__((ext_vector_type(4)))  u32    v4u;
typedef __attribute__((ext_vector_type(8)))  int    v8i;
typedef __attribute__((ext_vector_type(4)))  int    v4i;

#ifdef __has_builtin
#if __has_builtin(__builtin_amdgcn_tensor_load_to_lds) && \
    __has_builtin(__builtin_amdgcn_s_wait_tensorcnt)
#define HAVE_TDM 1
#endif
#endif
#ifndef HAVE_TDM
#define HAVE_TDM 0
#endif

union Frag { v16bf v; u32 u[8]; };

__device__ inline u16 f2bf(float f) {
    u32 x = __float_as_uint(f);
    return (u16)((x + 0x7fffu + ((x >> 16) & 1u)) >> 16);
}
__device__ inline u32 pack2bf(float a, float b) {
    return (u32)f2bf(a) | ((u32)f2bf(b) << 16);
}

__device__ inline v8f wmma32(const Frag& a, const Frag& b, v8f c) {
    return __builtin_amdgcn_wmma_f32_16x16x32_bf16(
        false, a.v, false, b.v, (short)0, c, false, false);
}

// A-fragment (16x32 bf16): lanes 0-15 K-half 0, lanes 16-31 K-half 1;
// pair index (v<4 ? v : v+4) + 4*half.
__device__ inline void loadA(Frag& f, const u32* rowBase, int hi16) {
    int o = 4 * hi16;
#pragma unroll
    for (int v = 0; v < 8; ++v)
        f.u[v] = rowBase[(v < 4 ? v : v + 4) + o];
}
// B-fragment (32x16 bf16) from Bt[N][K] rows: pairs (v<4 ? v : v+4) + 4*half.
__device__ inline void loadB(Frag& f, const u32* rowBase, int hi16) {
    int o = 4 * hi16;
#pragma unroll
    for (int v = 0; v < 8; ++v)
        f.u[v] = rowBase[(v < 4 ? v : v + 4) + o];
}

#if HAVE_TDM
// Issue a TDM 2-D tile load (bf16 elements) global -> LDS.
// tile_d0 = contiguous elements per row, tile_d1 = rows,
// row_stride = global row stride in elements.
// padI/padA encode LDS padding: pad_interval code (0=2..7=256 DWORDs),
// pad_amount code (n -> n+1 DWORDs inserted).
__device__ inline void tdm_load_2d(u32 lds_addr, const u16* gptr,
                                   u32 tile_d0, u32 tile_d1,
                                   u32 row_stride, u32 padI, u32 padA) {
    u64 ga = (u64)gptr;
    v4u g0;
    g0[0] = 1u;                                   // count=1, user mode
    g0[1] = lds_addr;                             // lds_addr
    g0[2] = (u32)ga;                              // global_addr[31:0]
    g0[3] = ((u32)(ga >> 32) & 0x01FFFFFFu) | 0x80000000u; // addr[56:32]|type=2
    v8i g1;
    g1[0] = (int)((1u << 16) |                    // data_size = 2 bytes
                  (1u << 20) |                    // pad_enable
                  (padI << 22) | (padA << 25));
    g1[1] = (int)((tile_d0 & 0xFFFFu) << 16);     // tensor_dim0[15:0]
    g1[2] = (int)(((tile_d0 >> 16) & 0xFFFFu) |
                  ((tile_d1 & 0xFFFFu) << 16));   // dim0 hi | tensor_dim1 lo
    g1[3] = (int)(((tile_d1 >> 16) & 0xFFFFu) |
                  ((tile_d0 & 0xFFFFu) << 16));   // dim1 hi | tile_dim0
    g1[4] = (int)(tile_d1 & 0xFFFFu);             // tile_dim1, tile_dim2=0
    g1[5] = (int)row_stride;                      // tensor_dim0_stride[31:0]
    g1[6] = 0;                                    // stride hi, dim1_stride lo
    g1[7] = 0;
    v4i z4 = {0, 0, 0, 0};
#if __clang_major__ >= 23
    v8i z8 = {0, 0, 0, 0, 0, 0, 0, 0};
    __builtin_amdgcn_tensor_load_to_lds(g0, g1, z4, z4, z8, 0);
#else
    __builtin_amdgcn_tensor_load_to_lds(g0, g1, z4, z4, 0);
#endif
}
#endif

// ---------------------------------------------------------------------------
// f32 -> bf16 elementwise convert (pairs)
// ---------------------------------------------------------------------------
__global__ __launch_bounds__(256) void convert_bf16_kernel(
    const float* __restrict__ in, u16* __restrict__ out, int npairs) {
    int i = blockIdx.x * 256 + threadIdx.x;
    if (i < npairs) {
        float2 f = ((const float2*)in)[i];
        ((u32*)out)[i] = pack2bf(f.x, f.y);
    }
}

// ---------------------------------------------------------------------------
// f32 [R][C] -> bf16 [C][R] transpose
// ---------------------------------------------------------------------------
__global__ __launch_bounds__(256) void transpose_bf16_kernel(
    const float* __restrict__ in, u16* __restrict__ out, int R, int C) {
    __shared__ float tile[32][33];
    int bx = blockIdx.x * 32, by = blockIdx.y * 32;
    int tx = threadIdx.x, ty = threadIdx.y;
#pragma unroll
    for (int i = 0; i < 4; ++i)
        tile[ty + i * 8][tx] = in[(size_t)(by + ty + i * 8) * C + bx + tx];
    __syncthreads();
#pragma unroll
    for (int i = 0; i < 4; ++i)
        out[(size_t)(bx + ty + i * 8) * R + by + tx] = f2bf(tile[tx][ty + i * 8]);
}

// ---------------------------------------------------------------------------
// Generic TN GEMM: C[M][N] = A_bf16[M][K] * Bt_bf16[N][K]  (f32 accum, WMMA)
// TDM double-buffered pipeline: wave0 issues tile k+1 via tensor DMA while
// all waves compute tile k from LDS.
// MODE 0: QKV epilogue; MODE 1: bias+ReLU->bf16; MODE 2: bias+residual->f32
// ---------------------------------------------------------------------------
template <int MODE>
__global__ __launch_bounds__(256) void gemm_bf16_kernel(
    const u16* __restrict__ A, const u16* __restrict__ Bt,
    const float* __restrict__ bias, const float* __restrict__ res,
    float* __restrict__ outf, u16* __restrict__ outb,
    u16* __restrict__ qb, u16* __restrict__ kb, u16* __restrict__ vb,
    int M, int N, int K) {
    constexpr int SP = 20; // uint pairs per LDS row (16 data + 4 pad)
#if HAVE_TDM
    __shared__ u32 As[2][128 * SP];
    __shared__ u32 Bs[2][128 * SP];
#else
    __shared__ u32 As[1][128 * SP];
    __shared__ u32 Bs[1][128 * SP];
#endif

    int tid = threadIdx.x;
    int lane = tid & 31, wave = tid >> 5;
    int hi16 = lane >> 4, l16 = lane & 15;
    int wm = wave >> 2, wn = wave & 3;           // 2 x 4 wave grid
    int m0 = blockIdx.y * 128, n0 = blockIdx.x * 128;

    v8f acc[4][2];
#pragma unroll
    for (int i = 0; i < 4; ++i)
#pragma unroll
        for (int j = 0; j < 2; ++j) acc[i][j] = (v8f){0, 0, 0, 0, 0, 0, 0, 0};

    int kTiles = K >> 5;

#if HAVE_TDM
    if (wave == 0) {
        tdm_load_2d((u32)(u64)&As[0][0], A + (size_t)m0 * K, 32, 128, (u32)K, 3, 3);
        tdm_load_2d((u32)(u64)&Bs[0][0], Bt + (size_t)n0 * K, 32, 128, (u32)K, 3, 3);
    }
#endif

    for (int kt = 0; kt < kTiles; ++kt) {
#if HAVE_TDM
        int cur = kt & 1;
        if (wave == 0) {
            if (kt + 1 < kTiles) {
                int ko = (kt + 1) << 5;
                tdm_load_2d((u32)(u64)&As[cur ^ 1][0],
                            A + (size_t)m0 * K + ko, 32, 128, (u32)K, 3, 3);
                tdm_load_2d((u32)(u64)&Bs[cur ^ 1][0],
                            Bt + (size_t)n0 * K + ko, 32, 128, (u32)K, 3, 3);
                __builtin_amdgcn_s_wait_tensorcnt((short)2); // tile kt complete
            } else {
                __builtin_amdgcn_s_wait_tensorcnt((short)0);
            }
        }
#else
        int cur = 0;
        int kofs = kt << 5;
#pragma unroll
        for (int i = 0; i < 2; ++i) {
            int idx = tid + i * 256;
            int r = idx >> 2, c = idx & 3;
            *(uint4*)&As[0][r * SP + c * 4] =
                *(const uint4*)(A + (size_t)(m0 + r) * K + kofs + c * 8);
            *(uint4*)&Bs[0][r * SP + c * 4] =
                *(const uint4*)(Bt + (size_t)(n0 + r) * K + kofs + c * 8);
        }
        if (kt + 1 < kTiles) {
            int r = tid >> 2, c = tid & 3;
            __builtin_prefetch(A + (size_t)(m0 + r) * K + kofs + 32 + c * 8, 0, 0);
            __builtin_prefetch(Bt + (size_t)(n0 + r) * K + kofs + 32 + c * 8, 0, 0);
        }
#endif
        __syncthreads();

        Frag a[4], b[2];
#pragma unroll
        for (int mt = 0; mt < 4; ++mt)
            loadA(a[mt], &As[cur][(wm * 64 + mt * 16 + l16) * SP], hi16);
#pragma unroll
        for (int nt = 0; nt < 2; ++nt)
            loadB(b[nt], &Bs[cur][(wn * 32 + nt * 16 + l16) * SP], hi16);
#pragma unroll
        for (int mt = 0; mt < 4; ++mt)
#pragma unroll
            for (int nt = 0; nt < 2; ++nt)
                acc[mt][nt] = wmma32(a[mt], b[nt], acc[mt][nt]);
        __syncthreads();
    }

#pragma unroll
    for (int mt = 0; mt < 4; ++mt)
#pragma unroll
        for (int nt = 0; nt < 2; ++nt)
#pragma unroll
            for (int r = 0; r < 8; ++r) {
                int row = m0 + wm * 64 + mt * 16 + r + 8 * hi16;
                int col = n0 + wn * 32 + nt * 16 + l16;
                float val = acc[mt][nt][r];
                if (MODE == 0) {
                    val += bias[col];
                    int head = col / 192;
                    int part = (col % 192) / 64;
                    int d = col % 64;
                    int bb = row >> 11, s = row & 2047;
                    size_t o = (((size_t)(bb * 16 + head)) * 2048 + s) * 64 + d;
                    if (part == 0)      qb[o] = f2bf(val * 0.125f); // fold 1/sqrt(64)
                    else if (part == 1) kb[o] = f2bf(val);
                    else                vb[o] = f2bf(val);
                } else if (MODE == 1) {
                    val += bias[col];
                    val = fmaxf(val, 0.0f);
                    outb[(size_t)row * N + col] = f2bf(val);
                } else {
                    val += bias[col] + res[(size_t)row * N + col];
                    outf[(size_t)row * N + col] = val;
                }
            }
}

// ---------------------------------------------------------------------------
// Flash-style attention: one block = (batch, head, 128 query rows)
// Q pre-scaled; S = Q K^T via WMMA, online softmax, O += P V via WMMA.
// Q and K tiles staged by TDM; V transposed manually into LDS.
// ---------------------------------------------------------------------------
__global__ __launch_bounds__(256) void attn_kernel(
    const u16* __restrict__ qg, const u16* __restrict__ kg,
    const u16* __restrict__ vg, float* __restrict__ outp) {
    __shared__ u32 Qs[128 * 36];     // 128 q rows x 32 d-pairs (+pad)
    __shared__ u32 Ks[64 * 36];      // 64 key rows x 32 d-pairs
    __shared__ u32 Vts[64 * 36];     // V^T: 64 d rows x 32 key-pairs
    __shared__ u32 Ps[8 * 16 * 36];  // per-wave P: 16 rows x 32 key-pairs

    int tid = threadIdx.x;
    int lane = tid & 31, w = tid >> 5;
    int hi16 = lane >> 4, l16 = lane & 15;
    int bid = blockIdx.x;
    int qi = bid & 15, head = (bid >> 4) & 15, batch = bid >> 8;

    size_t hofs = ((size_t)(batch * 16 + head)) * 2048 * 64;
    const u16* Q = qg + hofs + (size_t)qi * 128 * 64;
    const u16* Kp = kg + hofs;
    const u16* Vp = vg + hofs;

#if HAVE_TDM
    if (w == 0) tdm_load_2d((u32)(u64)&Qs[0], Q, 64, 128, 64, 4, 3);
#else
#pragma unroll
    for (int i = 0; i < 4; ++i) {
        int idx = tid + i * 256;
        int r = idx >> 3, c = idx & 7;
        *(uint4*)&Qs[r * 36 + c * 4] = *(const uint4*)(Q + r * 64 + c * 8);
    }
#endif

    float rowmax[8], rowsum[8];
    v8f oacc[4];
#pragma unroll
    for (int r = 0; r < 8; ++r) { rowmax[r] = -3.0e38f; rowsum[r] = 0.0f; }
#pragma unroll
    for (int dt = 0; dt < 4; ++dt) oacc[dt] = (v8f){0, 0, 0, 0, 0, 0, 0, 0};

    for (int kb0 = 0; kb0 < 2048; kb0 += 64) {
        // K chunk [64 keys][64 d]
#if HAVE_TDM
        if (w == 0) tdm_load_2d((u32)(u64)&Ks[0], Kp + (size_t)kb0 * 64,
                                64, 64, 64, 4, 3);
#else
#pragma unroll
        for (int i = 0; i < 2; ++i) {
            int idx = tid + i * 256;
            int r = idx >> 3, c = idx & 7;
            *(uint4*)&Ks[r * 36 + c * 4] =
                *(const uint4*)(Kp + (size_t)(kb0 + r) * 64 + c * 8);
        }
#endif
        // V chunk transposed into Vts[d][keys]
        u16* Vus = (u16*)Vts;
#pragma unroll
        for (int i = 0; i < 8; ++i) {
            int idx = tid + i * 256;
            int key = idx >> 5, cp = idx & 31;
            u32 u = *(const u32*)(Vp + (size_t)(kb0 + key) * 64 + cp * 2);
            Vus[(cp * 2) * 72 + key] = (u16)(u & 0xffffu);
            Vus[(cp * 2 + 1) * 72 + key] = (u16)(u >> 16);
        }
        if (kb0 + 64 < 2048) {
            int r = tid >> 3;
            __builtin_prefetch(Kp + (size_t)(kb0 + 64 + r) * 64, 0, 0);
            __builtin_prefetch(Vp + (size_t)(kb0 + 64 + r) * 64, 0, 0);
        }
#if HAVE_TDM
        if (w == 0) __builtin_amdgcn_s_wait_tensorcnt((short)0); // Q + K done
#endif
        __syncthreads();

        // S(16 x 64) = Q_wave K^T, accumulated over d (two 32-chunks)
        v8f sacc[4];
#pragma unroll
        for (int t = 0; t < 4; ++t) sacc[t] = (v8f){0, 0, 0, 0, 0, 0, 0, 0};
#pragma unroll
        for (int kk = 0; kk < 2; ++kk) {
            Frag aq;
            loadA(aq, &Qs[(w * 16 + l16) * 36 + kk * 16], hi16);
#pragma unroll
            for (int t = 0; t < 4; ++t) {
                Frag bk;
                loadB(bk, &Ks[(t * 16 + l16) * 36 + kk * 16], hi16);
                sacc[t] = wmma32(aq, bk, sacc[t]);
            }
        }

        // online softmax (rows r + 8*hi16; 16-lane reductions)
        u16* Pus = (u16*)Ps;
#pragma unroll
        for (int r = 0; r < 8; ++r) {
            float m = fmaxf(fmaxf(sacc[0][r], sacc[1][r]),
                            fmaxf(sacc[2][r], sacc[3][r]));
#pragma unroll
            for (int s = 1; s < 16; s <<= 1) m = fmaxf(m, __shfl_xor(m, s, 32));
            float nm = fmaxf(rowmax[r], m);
            float sc = __expf(rowmax[r] - nm);
            rowmax[r] = nm;
            float ps = 0.0f;
#pragma unroll
            for (int t = 0; t < 4; ++t) {
                float p = __expf(sacc[t][r] - nm);
                sacc[t][r] = p;
                ps += p;
            }
#pragma unroll
            for (int s = 1; s < 16; s <<= 1) ps += __shfl_xor(ps, s, 32);
            rowsum[r] = rowsum[r] * sc + ps;
#pragma unroll
            for (int dt = 0; dt < 4; ++dt) oacc[dt][r] *= sc;
#pragma unroll
            for (int t = 0; t < 4; ++t)
                Pus[(w * 16 + r + 8 * hi16) * 72 + t * 16 + l16] = f2bf(sacc[t][r]);
        }

        // O(16 x 64) += P(16 x 64keys) V(64keys x 64)
#pragma unroll
        for (int kk = 0; kk < 2; ++kk) {
            Frag ap;
            loadA(ap, &Ps[(w * 16 + l16) * 36 + kk * 16], hi16);
#pragma unroll
            for (int dt = 0; dt < 4; ++dt) {
                Frag bv;
                loadB(bv, &Vts[(dt * 16 + l16) * 36 + kk * 16], hi16);
                oacc[dt] = wmma32(ap, bv, oacc[dt]);
            }
        }
        __syncthreads();
    }

    // store in reference's faithful (b,h,s,d) flat order
    float* O = outp + (size_t)batch * 2048 * 1024 + (size_t)head * 2048 * 64;
    int s0 = qi * 128 + w * 16;
#pragma unroll
    for (int r = 0; r < 8; ++r) {
        float inv = 1.0f / rowsum[r];
        int s = s0 + r + 8 * hi16;
#pragma unroll
        for (int dt = 0; dt < 4; ++dt)
            O[(size_t)s * 64 + dt * 16 + l16] = oacc[dt][r] * inv;
    }
}

// ---------------------------------------------------------------------------
// Fused residual + LayerNorm; one block per 1024-wide row.
// ---------------------------------------------------------------------------
__global__ __launch_bounds__(256) void ln_kernel(
    const float* __restrict__ a, const float* __restrict__ res,
    const float* __restrict__ gamma, const float* __restrict__ beta,
    float* __restrict__ outf, u16* __restrict__ outb) {
    int row = blockIdx.x, tid = threadIdx.x;
    size_t base = (size_t)row * 1024;
    float4 v = *(const float4*)(a + base + tid * 4);
    if (res) {
        float4 rr = *(const float4*)(res + base + tid * 4);
        v.x += rr.x; v.y += rr.y; v.z += rr.z; v.w += rr.w;
    }
    float s = v.x + v.y + v.z + v.w;
    float q = v.x * v.x + v.y * v.y + v.z * v.z + v.w * v.w;
#pragma unroll
    for (int m = 1; m < 32; m <<= 1) {
        s += __shfl_xor(s, m, 32);
        q += __shfl_xor(q, m, 32);
    }
    __shared__ float rs[8], rq[8];
    int w = tid >> 5;
    if ((tid & 31) == 0) { rs[w] = s; rq[w] = q; }
    __syncthreads();
    s = 0.0f; q = 0.0f;
#pragma unroll
    for (int i = 0; i < 8; ++i) { s += rs[i]; q += rq[i]; }
    float mean = s * (1.0f / 1024.0f);
    float var = q * (1.0f / 1024.0f) - mean * mean;
    float rstd = rsqrtf(var + 1e-5f);
    float4 g = *(const float4*)(gamma + tid * 4);
    float4 bt = *(const float4*)(beta + tid * 4);
    float4 o;
    o.x = g.x * ((v.x - mean) * rstd) + bt.x;
    o.y = g.y * ((v.y - mean) * rstd) + bt.y;
    o.z = g.z * ((v.z - mean) * rstd) + bt.z;
    o.w = g.w * ((v.w - mean) * rstd) + bt.w;
    *(float4*)(outf + base + tid * 4) = o;
    if (outb) {
        size_t c = base + tid * 4;
        *(u32*)&outb[c] = pack2bf(o.x, o.y);
        *(u32*)&outb[c + 2] = pack2bf(o.z, o.w);
    }
}

// ---------------------------------------------------------------------------
extern "C" void kernel_launch(void* const* d_in, const int* in_sizes, int n_in,
                              void* d_out, int out_size, void* d_ws, size_t ws_size,
                              hipStream_t stream) {
    const int D = 1024, F = 4096, M = 4096, N3 = 3072; // M = B*S

    const float* x     = (const float*)d_in[0];
    const float* Wqkv  = (const float*)d_in[1];
    const float* bqkv  = (const float*)d_in[2];
    const float* gamma = (const float*)d_in[3];
    const float* beta  = (const float*)d_in[4];
    const float* W1    = (const float*)d_in[5];
    const float* b1    = (const float*)d_in[6];
    const float* W2    = (const float*)d_in[7];
    const float* b2    = (const float*)d_in[8];
    float* out = (float*)d_out;

    size_t off = 0;
    auto alloc = [&](size_t bytes) -> void* {
        void* p = (char*)d_ws + off;
        off += (bytes + 255) & ~(size_t)255;
        return p;
    };
    u16* xb    = (u16*)alloc((size_t)M * D * 2);
    u16* wqkvT = (u16*)alloc((size_t)N3 * D * 2);
    u16* w1T   = (u16*)alloc((size_t)F * D * 2);
    u16* w2T   = (u16*)alloc((size_t)D * F * 2);
    u16* qbuf  = (u16*)alloc((size_t)M * D * 2);
    u16* kbuf  = (u16*)alloc((size_t)M * D * 2);
    u16* vbuf  = (u16*)alloc((size_t)M * D * 2);
    float* attn = (float*)alloc((size_t)M * D * 4);
    float* hf   = (float*)alloc((size_t)M * D * 4);
    u16*   hb   = (u16*)alloc((size_t)M * D * 2);
    u16*   f1b  = (u16*)alloc((size_t)M * F * 2);

    // 1) convert x -> bf16; transpose-convert weights to [N][K] bf16
    convert_bf16_kernel<<<(M * D / 2 + 255) / 256, 256, 0, stream>>>(x, xb, M * D / 2);
    transpose_bf16_kernel<<<dim3(N3 / 32, D / 32), dim3(32, 8), 0, stream>>>(Wqkv, wqkvT, D, N3);
    transpose_bf16_kernel<<<dim3(F / 32, D / 32), dim3(32, 8), 0, stream>>>(W1, w1T, D, F);
    transpose_bf16_kernel<<<dim3(D / 32, F / 32), dim3(32, 8), 0, stream>>>(W2, w2T, F, D);

    // 2) QKV projection (scatter into per-head Q/K/V, Q pre-scaled)
    gemm_bf16_kernel<0><<<dim3(N3 / 128, M / 128), 256, 0, stream>>>(
        xb, wqkvT, bqkv, nullptr, nullptr, nullptr, qbuf, kbuf, vbuf, M, N3, D);

    // 3) attention (2 batches x 16 heads x 16 query-blocks)
    attn_kernel<<<512, 256, 0, stream>>>(qbuf, kbuf, vbuf, attn);

    // 4) h = LN(x + attn_out)  (f32 + bf16 copies)
    ln_kernel<<<M, 256, 0, stream>>>(attn, x, gamma, beta, hf, hb);

    // 5) FFN1: relu(h @ W1 + b1) -> bf16
    gemm_bf16_kernel<1><<<dim3(F / 128, M / 128), 256, 0, stream>>>(
        hb, w1T, b1, nullptr, nullptr, f1b, nullptr, nullptr, nullptr, M, F, D);

    // 6) FFN2: f = relu_out @ W2 + b2 + h  -> d_out (f32)
    gemm_bf16_kernel<2><<<dim3(D / 128, M / 128), 256, 0, stream>>>(
        f1b, w2T, b2, hf, out, nullptr, nullptr, nullptr, nullptr, M, D, F);

    // 7) out = LN(f) in place
    ln_kernel<<<M, 256, 0, stream>>>(out, nullptr, gamma, beta, out, nullptr);
}
